// WindowAttention_55310588838457
// MI455X (gfx1250) — matrix operand order repaired and checked by reading, not compiled, not run on hardware
//
#include <hip/hip_runtime.h>
#include <hip/hip_bf16.h>

typedef _Float16 h4  __attribute__((ext_vector_type(4)));
typedef _Float16 h8  __attribute__((ext_vector_type(8)));
typedef _Float16 v16h __attribute__((ext_vector_type(16)));
typedef float    v8f  __attribute__((ext_vector_type(8)));

#define N_TOK 144
#define DIMC  384
#define NH    12
#define DH    32
#define KPAD  160
#define QKV3  1152

// ---------- WMMA fragment loaders (wave32, per ISA 7.12.2 layouts) ----------
// A-matrix 16x32 f16: lanes 0-15 hold M=lane, K = {0..7, 16..23}; lanes 16-31
// hold M=lane-16, K = {8..15, 24..31}.
__device__ __forceinline__ v16h load_a(const _Float16* base, int row0, int k0, int stride) {
    const int lane = threadIdx.x & 31;
    const int l = lane & 15, hs = lane >> 4;
    const _Float16* p = base + (size_t)(row0 + l) * stride + k0 + hs * 8;
    h8 lo = *(const h8*)(p);
    h8 hi = *(const h8*)(p + 16);
    v16h r;
#pragma unroll
    for (int i = 0; i < 8; ++i) { r[i] = lo[i]; r[i + 8] = hi[i]; }
    return r;
}

// B-matrix 32x16 f16 (stored N-major, i.e. Bt[n][k]): lanes 0-15 hold N=lane,
// K = {0..7, 16..23}; lanes 16-31 hold N=lane-16, K = {8..15, 24..31}.
__device__ __forceinline__ v16h load_b(const _Float16* base, int n0, int k0, int stride) {
    const int lane = threadIdx.x & 31;
    const int l = lane & 15, hs = lane >> 4;
    const _Float16* p = base + (size_t)(n0 + l) * stride + k0 + hs * 8;
    h8 lo = *(const h8*)(p);
    h8 hi = *(const h8*)(p + 16);
    v16h r;
#pragma unroll
    for (int i = 0; i < 8; ++i) { r[i] = lo[i]; r[i + 8] = hi[i]; }
    return r;
}

#define WMMA_F16(a, b, c) \
    __builtin_amdgcn_wmma_f32_16x16x32_f16(false, (a), false, (b), (short)0, (c), false, false)

// ---------- async global->LDS copy (CDNA5 GLOBAL_LOAD_ASYNC_TO_LDS_B128) ----------
__device__ __forceinline__ void async_copy_b128(unsigned lds_addr, const void* gptr) {
    asm volatile("global_load_async_to_lds_b128 %0, %1, off"
                 :: "v"(lds_addr), "v"((unsigned long long)(uintptr_t)gptr)
                 : "memory");
}
__device__ __forceinline__ void wait_async0() {
    asm volatile("s_wait_asynccnt 0x0" ::: "memory");
}

// ---------- Phase -1: x f32 -> f16 (once; enables async staging + halves re-reads) ----------
__global__ __launch_bounds__(256) void cvt_x_kernel(const float* __restrict__ x,
                                                    _Float16* __restrict__ xh, long long n4) {
    long long i = ((long long)blockIdx.x * 256 + threadIdx.x);
    if (i >= n4) return;
    const float4 v = *(const float4*)(x + i * 4);
    h4 o; o[0] = (_Float16)v.x; o[1] = (_Float16)v.y; o[2] = (_Float16)v.z; o[3] = (_Float16)v.w;
    *(h4*)(xh + i * 4) = o;
}

// ---------- Phase 0a: CPB MLP -> bias table (529 x 12) ----------
__global__ void cpb_kernel(const float* __restrict__ w1, const float* __restrict__ b1,
                           const float* __restrict__ w2, float* __restrict__ bias_table) {
    int t = blockIdx.x * blockDim.x + threadIdx.x;
    if (t >= 23 * 23) return;
    int i = t / 23, j = t - i * 23;
    float ch = (float)(i - 11) * (8.0f / 11.0f);
    float cw = (float)(j - 11) * (8.0f / 11.0f);
    const float inv_l2_8 = 1.0f / 3.0f;
    float t0 = (ch > 0.f ? 1.f : (ch < 0.f ? -1.f : 0.f)) * log2f(fabsf(ch) + 1.0f) * inv_l2_8;
    float t1 = (cw > 0.f ? 1.f : (cw < 0.f ? -1.f : 0.f)) * log2f(fabsf(cw) + 1.0f) * inv_l2_8;
    float acc[NH];
#pragma unroll
    for (int hh = 0; hh < NH; ++hh) acc[hh] = 0.f;
    for (int u = 0; u < 512; ++u) {
        float hu = fmaxf(t0 * w1[u] + t1 * w1[512 + u] + b1[u], 0.0f);
#pragma unroll
        for (int hh = 0; hh < NH; ++hh) acc[hh] += hu * w2[u * NH + hh];
    }
#pragma unroll
    for (int hh = 0; hh < NH; ++hh) bias_table[t * NH + hh] = acc[hh];
}

// ---------- Phase 0b: expand 16*sigmoid(bias) to (H, N, N) ----------
__global__ void relbias_kernel(const float* __restrict__ bias_table, float* __restrict__ rel) {
    int idx = blockIdx.x * 256 + threadIdx.x;
    if (idx >= NH * N_TOK * N_TOK) return;
    int h = idx / (N_TOK * N_TOK);
    int r = idx - h * N_TOK * N_TOK;
    int i = r / N_TOK, j = r - i * N_TOK;
    int ih = i / 12, iw = i - ih * 12;
    int jh = j / 12, jw = j - jh * 12;
    int rp = (ih - jh + 11) * 23 + (iw - jw + 11);
    float v = bias_table[rp * NH + h];
    rel[idx] = 16.0f / (1.0f + __expf(-v));
}

// ---------- Phase 1: fused QKV + cosine attention + softmax + AV ----------
struct alignas(16) SharedA {
    union { _Float16 xs[N_TOK * DIMC]; float attn[N_TOK * N_TOK]; } u1;  // 110592 B
    union { _Float16 wt[96 * DIMC]; _Float16 attnh[N_TOK * KPAD]; } u2;  //  73728 B
    float    qf[N_TOK * DH];                                             //  18432 B
    float    kf[N_TOK * DH];                                             //  18432 B
    _Float16 qh[N_TOK * DH];                                             //   9216 B
    _Float16 kh[N_TOK * DH];                                             //   9216 B
    _Float16 vt[DH * KPAD];                                              //  10240 B
};                                                                       // 249856 B total

__global__ __launch_bounds__(256) void attn_kernel(
    const _Float16* __restrict__ xh, const float* __restrict__ w_qkv,
    const float* __restrict__ q_bias, const float* __restrict__ v_bias,
    const float* __restrict__ logit_scale, const float* __restrict__ rel_bias,
    _Float16* __restrict__ ctx)
{
    __shared__ SharedA sm;
    const int h   = blockIdx.x;     // head
    const int win = blockIdx.y;     // window
    const int tid = threadIdx.x;
    const int wid = tid >> 5;
    const int lane = tid & 31;
    const int l = lane & 15, hs = lane >> 4;

    // Async-stage x tile (f16, 110592 B) into LDS -- overlaps with wt staging below.
    {
        const _Float16* g = xh + (size_t)win * N_TOK * DIMC;
        unsigned lds_base = (unsigned)(uintptr_t)(void*)&sm.u1.xs[0];
        for (int it = tid; it < (N_TOK * DIMC) / 8; it += 256)   // 27 iters/thread
            async_copy_b128(lds_base + it * 16, g + it * 8);
    }
    // Stage per-head q|k|v weight slices, transposed to N-major (96 x 384), f32->f16.
    for (int i = tid; i < 96 * DIMC; i += 256) {
        int n = i / DIMC, k = i - n * DIMC;
        int sec = n >> 5, c = n & 31;
        int col = sec * DIMC + h * DH + c;
        sm.u2.wt[i] = (_Float16)w_qkv[(size_t)k * QKV3 + col];
    }
    wait_async0();
    __syncthreads();

    // ---- Stage A: [144x384] @ [384x96] -> q_raw | k_raw | v_raw ----
    // 27 groups of (1 A-tile x 2 B-tiles): A fragment reused across 2 WMMAs.
    for (int g = wid; g < 27; g += 8) {
        int mt = g / 3, np = g - mt * 3;          // np == section (q/k/v)
        v8f acc0 = {}, acc1 = {};
        for (int kk = 0; kk < 12; ++kk) {
            v16h a  = load_a(sm.u1.xs, mt * 16, kk * 32, DIMC);
            v16h b0 = load_b(sm.u2.wt, (np * 2 + 0) * 16, kk * 32, DIMC);
            v16h b1 = load_b(sm.u2.wt, (np * 2 + 1) * 16, kk * 32, DIMC);
            acc0 = WMMA_F16(a, b0, acc0);
            acc1 = WMMA_F16(a, b1, acc1);
        }
        int m0 = mt * 16 + hs * 8;
        if (np == 0) {
            float bq0 = q_bias[h * DH + l], bq1 = q_bias[h * DH + 16 + l];
#pragma unroll
            for (int j = 0; j < 8; ++j) {
                sm.qf[(m0 + j) * DH + l]      = acc0[j] + bq0;
                sm.qf[(m0 + j) * DH + 16 + l] = acc1[j] + bq1;
            }
        } else if (np == 1) {
#pragma unroll
            for (int j = 0; j < 8; ++j) {
                sm.kf[(m0 + j) * DH + l]      = acc0[j];
                sm.kf[(m0 + j) * DH + 16 + l] = acc1[j];
            }
        } else {
            float bv0 = v_bias[h * DH + l], bv1 = v_bias[h * DH + 16 + l];
#pragma unroll
            for (int j = 0; j < 8; ++j) {      // transposed v^T store
                sm.vt[l * KPAD + (m0 + j)]        = (_Float16)(acc0[j] + bv0);
                sm.vt[(16 + l) * KPAD + (m0 + j)] = (_Float16)(acc1[j] + bv1);
            }
        }
    }
    // zero-pad v^T K-tail rows 144..159
    for (int i = tid; i < DH * 16; i += 256) {
        int c = i >> 4, r = i & 15;
        sm.vt[c * KPAD + N_TOK + r] = (_Float16)0.0f;
    }
    __syncthreads();

    // ---- l2-normalize rows of q,k; fold clamped temperature into q ----
    float scale = __expf(fminf(logit_scale[h], 4.605170185988092f));  // ln(100)
    for (int r = tid; r < 2 * N_TOK; r += 256) {
        int row = (r < N_TOK) ? r : r - N_TOK;
        const float* src = (r < N_TOK) ? sm.qf : sm.kf;
        float ss = 0.f;
#pragma unroll
        for (int c = 0; c < DH; ++c) { float v = src[row * DH + c]; ss += v * v; }
        float rn = rsqrtf(fmaxf(ss, 1e-12f));
        if (r < N_TOK) {
            rn *= scale;
#pragma unroll
            for (int c = 0; c < DH; ++c) sm.qh[row * DH + c] = (_Float16)(sm.qf[row * DH + c] * rn);
        } else {
#pragma unroll
            for (int c = 0; c < DH; ++c) sm.kh[row * DH + c] = (_Float16)(sm.kf[row * DH + c] * rn);
        }
    }
    __syncthreads();

    // ---- Stage B: attn = qn @ kn^T + rel_bias (K=32): A reused across 3 WMMAs ----
    const float* rb = rel_bias + (size_t)h * N_TOK * N_TOK;
    for (int g = wid; g < 27; g += 8) {
        int mt = g / 3, nb = g - mt * 3;
        v16h a = load_a(sm.qh, mt * 16, 0, DH);
        int m0 = mt * 16 + hs * 8;
#pragma unroll
        for (int q = 0; q < 3; ++q) {
            int nt = nb * 3 + q;
            v16h b = load_b(sm.kh, nt * 16, 0, DH);
            v8f acc = {};
            acc = WMMA_F16(a, b, acc);
            int n = nt * 16 + l;
#pragma unroll
            for (int j = 0; j < 8; ++j)
                sm.u1.attn[(m0 + j) * N_TOK + n] = acc[j] + rb[(size_t)(m0 + j) * N_TOK + n];
        }
    }
    __syncthreads();

    // ---- row softmax (f32), emit f16 padded to K=160 ----
    for (int row = tid; row < N_TOK; row += 256) {
        float* ar = &sm.u1.attn[row * N_TOK];
        float mx = -1e30f;
        for (int j = 0; j < N_TOK; ++j) mx = fmaxf(mx, ar[j]);
        float s = 0.f;
        for (int j = 0; j < N_TOK; ++j) { float e = __expf(ar[j] - mx); ar[j] = e; s += e; }
        float inv = 1.0f / s;
        _Float16* hr = &sm.u2.attnh[row * KPAD];
        for (int j = 0; j < N_TOK; ++j) hr[j] = (_Float16)(ar[j] * inv);
        for (int j = N_TOK; j < KPAD; ++j) hr[j] = (_Float16)0.0f;
    }
    __syncthreads();

    // ---- Stage C: out_head = attn @ v (K=160, zero-padded): A reused x2 ----
    _Float16* cb = ctx + ((size_t)win * N_TOK) * DIMC + h * DH;
    for (int g = wid; g < 9; g += 8) {
        v8f acc0 = {}, acc1 = {};
        for (int kk = 0; kk < 5; ++kk) {
            v16h a  = load_a(sm.u2.attnh, g * 16, kk * 32, KPAD);
            v16h b0 = load_b(sm.vt, 0,  kk * 32, KPAD);
            v16h b1 = load_b(sm.vt, 16, kk * 32, KPAD);
            acc0 = WMMA_F16(a, b0, acc0);
            acc1 = WMMA_F16(a, b1, acc1);
        }
        int m0 = g * 16 + hs * 8;
#pragma unroll
        for (int j = 0; j < 8; ++j) {
            cb[(size_t)(m0 + j) * DIMC + l]      = (_Float16)acc0[j];
            cb[(size_t)(m0 + j) * DIMC + 16 + l] = (_Float16)acc1[j];
        }
    }
}

// ---------- Phase 2: output projection  ctx @ proj_w + proj_b ----------
struct alignas(16) SharedP {
    _Float16 xs[N_TOK * DIMC];   // 110592 B
    _Float16 pwt[96 * DIMC];     //  73728 B
};

__global__ __launch_bounds__(256) void proj_kernel(
    const _Float16* __restrict__ ctx, const float* __restrict__ proj_w,
    const float* __restrict__ proj_b, float* __restrict__ out)
{
    __shared__ SharedP sm;
    const int win = blockIdx.x;
    const int tid = threadIdx.x;
    const int wid = tid >> 5;
    const int lane = tid & 31;
    const int l = lane & 15, hs = lane >> 4;

    // Async-stage ctx tile (f16) into LDS; overlaps with first weight-chunk staging.
    {
        const _Float16* g = ctx + (size_t)win * N_TOK * DIMC;
        unsigned lds_base = (unsigned)(uintptr_t)(void*)&sm.xs[0];
        for (int it = tid; it < (N_TOK * DIMC) / 8; it += 256)
            async_copy_b128(lds_base + it * 16, g + it * 8);
    }
    float* ob = out + (size_t)win * N_TOK * DIMC;

    for (int chunk = 0; chunk < 4; ++chunk) {
        int n0 = chunk * 96;
        __syncthreads();  // previous chunk's pwt readers done
        for (int i = tid; i < 96 * DIMC; i += 256) {
            int n = i / DIMC, k = i - n * DIMC;
            sm.pwt[i] = (_Float16)proj_w[(size_t)k * DIMC + n0 + n];  // N-major
        }
        if (chunk == 0) wait_async0();
        __syncthreads();
        // 27 groups of (1 A-tile x 2 B-tiles)
        for (int g = wid; g < 27; g += 8) {
            int mt = g / 3, np = g - mt * 3;
            v8f acc0 = {}, acc1 = {};
            for (int kk = 0; kk < 12; ++kk) {
                v16h a  = load_a(sm.xs, mt * 16, kk * 32, DIMC);
                v16h b0 = load_b(sm.pwt, (np * 2 + 0) * 16, kk * 32, DIMC);
                v16h b1 = load_b(sm.pwt, (np * 2 + 1) * 16, kk * 32, DIMC);
                acc0 = WMMA_F16(a, b0, acc0);
                acc1 = WMMA_F16(a, b1, acc1);
            }
            int m0 = mt * 16 + hs * 8;
            int c0 = n0 + np * 32 + l, c1 = c0 + 16;
            float bb0 = proj_b[c0], bb1 = proj_b[c1];
#pragma unroll
            for (int j = 0; j < 8; ++j) {
                ob[(size_t)(m0 + j) * DIMC + c0] = acc0[j] + bb0;
                ob[(size_t)(m0 + j) * DIMC + c1] = acc1[j] + bb1;
            }
        }
    }
}

// ---------- host launcher ----------
extern "C" void kernel_launch(void* const* d_in, const int* in_sizes, int n_in,
                              void* d_out, int out_size, void* d_ws, size_t ws_size,
                              hipStream_t stream) {
    const float* x           = (const float*)d_in[0];
    const float* w_qkv       = (const float*)d_in[1];
    const float* q_bias      = (const float*)d_in[2];
    const float* v_bias      = (const float*)d_in[3];
    const float* logit_scale = (const float*)d_in[4];
    const float* cpb_w1      = (const float*)d_in[5];
    const float* cpb_b1      = (const float*)d_in[6];
    const float* cpb_w2      = (const float*)d_in[7];
    const float* proj_w      = (const float*)d_in[8];
    const float* proj_b      = (const float*)d_in[9];
    float* out = (float*)d_out;

    char* ws = (char*)d_ws;
    float*    bias_table = (float*)ws;                                   //     25,392 B
    float*    rel        = (float*)(ws + 32768);                         //    995,328 B
    _Float16* ctx        = (_Float16*)(ws + 32768 + 1048576);            // 56,623,104 B
    _Float16* xh         = (_Float16*)(ws + 32768 + 1048576 + 56623104); // 56,623,104 B

    const long long n4 = (long long)512 * N_TOK * DIMC / 4;              // 7,077,888
    cvt_x_kernel<<<(unsigned)((n4 + 255) / 256), 256, 0, stream>>>(x, xh, n4);
    cpb_kernel<<<3, 256, 0, stream>>>(cpb_w1, cpb_b1, cpb_w2, bias_table);
    relbias_kernel<<<(NH * N_TOK * N_TOK + 255) / 256, 256, 0, stream>>>(bias_table, rel);
    attn_kernel<<<dim3(NH, 512), 256, 0, stream>>>(xh, w_qkv, q_bias, v_bias,
                                                   logit_scale, rel, ctx);
    proj_kernel<<<512, 256, 0, stream>>>(ctx, proj_w, proj_b, out);
}
